// ScannedRNN_15479062134957
// MI455X (gfx1250) — matrix operand order repaired and checked by reading, not compiled, not run on hardware
//
#include <hip/hip_runtime.h>
#include <hip/hip_bf16.h>
#include <stdint.h>

// ---------------------------------------------------------------------------
// GRU scan for MI455X (gfx1250, wave32, WMMA).
//   Phase 0: transpose+convert Wi/Wh (f32 row-major [K,3H]) -> bf16 [3H,K].
//   Phase 1: gi = X @ Wi, bf16 WMMA GEMM. Each block stages its 128-column
//            WiT strip in LDS once; its 8 waves (8 m-tiles) share it.
//            gi stored tile-interleaved [m_tile][col][row16] for b128 reads.
//   Phase 2: sequential scan. 16 independent workgroups (one per 16-row
//            batch tile, WMMA M=16). 304 KB of Wh^T in LDS (36/48 tiles);
//            12 tiles stream from L2 ahead of the LDS-fed WMMAs. gi loads
//            issued before the GEMM complete under it. HW trans ops for
//            gates; biases folded into WMMA accumulator init.
// ---------------------------------------------------------------------------

typedef __bf16 bf16;
typedef __attribute__((ext_vector_type(16))) __bf16 v16bf;
typedef __attribute__((ext_vector_type(8)))  float  v8f;

#define T_STEPS 1024
#define BATCH   256
#define DIM     256
#define HID     256
#define NG      768
#define TILE_E  (NG * 16)   // elements per (t, group) gi tile = 12288

#define LDS_TILES  36    // Wh tiles resident in LDS (phase 2)
#define COL_STRIDE 264   // padded k-stride (elements), 132 dw = 4 mod 64
#define H_STRIDE   260   // hsh row stride (f32), 260 dw = 4 mod 64

__device__ __forceinline__ v16bf load_b_frag(const bf16* p) {
  union { int4 u[2]; v16bf v; } t;
  t.u[0] = *(const int4*)(p);
  t.u[1] = *(const int4*)(p + 8);
  return t.v;
}

__device__ __forceinline__ float fast_exp2(float x) {
#if __has_builtin(__builtin_amdgcn_exp2f)
  return __builtin_amdgcn_exp2f(x);
#else
  return exp2f(x);
#endif
}
__device__ __forceinline__ float fast_rcp(float x) {
#if __has_builtin(__builtin_amdgcn_rcpf)
  return __builtin_amdgcn_rcpf(x);
#else
  return 1.0f / x;
#endif
}
__device__ __forceinline__ float sigmoidf(float x) {
  return fast_rcp(1.0f + fast_exp2(x * -1.4426950408889634f));
}
__device__ __forceinline__ float fast_tanh(float x) {
#if __has_builtin(__builtin_amdgcn_tanhf)
  return __builtin_amdgcn_tanhf(x);
#else
  // tanh(x) = 1 - 2/(1 + exp2(2x*log2e))
  return 1.0f - 2.0f * fast_rcp(1.0f + fast_exp2(x * 2.8853900817779268f));
#endif
}

__device__ __forceinline__ v8f splat8(float v) {
  v8f s;
#pragma unroll
  for (int e = 0; e < 8; ++e) s[e] = v;
  return s;
}

// ---------------------------------------------------------------------------
// Phase 0: WiT[n][k] = bf16(Wi[k][n]); WhT[n][k] = bf16(Wh[k][n])
// ---------------------------------------------------------------------------
__global__ void convert_weights(const float* __restrict__ Wi,
                                const float* __restrict__ Wh,
                                bf16* __restrict__ WiT,
                                bf16* __restrict__ WhT) {
  const int n = blockIdx.x;
  const int k = threadIdx.x;
  WiT[(size_t)n * DIM + k] = (bf16)Wi[(size_t)k * NG + n];
  WhT[(size_t)n * DIM + k] = (bf16)Wh[(size_t)k * NG + n];
}

// ---------------------------------------------------------------------------
// Phase 1: gi = X @ Wi. Block = 8 waves sharing one 128-col n-strip (staged
// in LDS); wave w computes m-tile m_block*8 + w. Grid: 6 * 2048 = 12288.
// ---------------------------------------------------------------------------
__global__ void __launch_bounds__(256) input_gemm(const float* __restrict__ X,
                                                  const bf16* __restrict__ WiT,
                                                  bf16* __restrict__ gi) {
  __shared__ __align__(16) bf16 wi_lds[128 * COL_STRIDE];  // 67,584 B

  const int tid  = threadIdx.x;
  const int wave = tid >> 5;
  const int lane = tid & 31;
  const int l    = lane & 15;
  const int hi   = lane >> 4;

  const int n_strip = blockIdx.x % 6;
  const int m_tile  = (blockIdx.x / 6) * 8 + wave;

  // stage this strip's WiT columns into LDS (128 cols x 256 k, bf16)
  for (int u = tid; u < 128 * 16; u += 256) {
    const int col = u >> 4, kb = u & 15;
    const bf16* src = WiT + (size_t)(n_strip * 128 + col) * DIM + kb * 16;
    bf16* dst = &wi_lds[col * COL_STRIDE + kb * 16];
    *(int4*)dst       = *(const int4*)src;
    *(int4*)(dst + 8) = *(const int4*)(src + 8);
  }
  __syncthreads();

  const float* arow = X + (size_t)(m_tile * 16 + l) * DIM;

  v8f acc[8];
#pragma unroll
  for (int i = 0; i < 8; ++i) acc[i] = splat8(0.0f);

  for (int k0 = 0; k0 < DIM; k0 += 32) {
    const float* a0 = arow + k0 + hi * 16;
    const float* a1 = arow + k0 + hi * 16 + 8;
    v16bf af;
#pragma unroll
    for (int e = 0; e < 8; ++e) {
      af[e]     = (bf16)a0[e];
      af[8 + e] = (bf16)a1[e];
    }
#pragma unroll
    for (int nt = 0; nt < 8; ++nt) {
      const bf16* bp = &wi_lds[(nt * 16 + l) * COL_STRIDE + k0 + hi * 16];
      acc[nt] = __builtin_amdgcn_wmma_f32_16x16x32_bf16(
          false, af, false, load_b_frag(bp), (short)0, acc[nt], false, false);
    }
  }

  // C layout: VGPR r, lane(l,hi) -> (row = hi*8 + r, col = base + l).
  bf16* gp = gi + ((size_t)m_tile * NG + n_strip * 128 + l) * 16 + hi * 8;
#pragma unroll
  for (int nt = 0; nt < 8; ++nt) {
    union { int4 u; bf16 h[8]; } pk;
#pragma unroll
    for (int r = 0; r < 8; ++r) pk.h[r] = (bf16)acc[nt][r];
    *(int4*)(gp + (size_t)nt * 256) = pk.u;
  }
}

// ---------------------------------------------------------------------------
// Phase 2: sequential scan. Block g owns batch rows [16g, 16g+16).
// Wave w owns hidden columns [32w, 32w+32): 6 accum tiles (r,z,n) x 2.
// j6<4 -> LDS; j6==4 -> LDS for waves 4..7, streamed for 0..3; j6==5 streamed.
// ---------------------------------------------------------------------------
__global__ void __launch_bounds__(256, 1) rnn_scan(
    const bf16* __restrict__ gi, const bf16* __restrict__ WhT,
    const float* __restrict__ bias, const uint8_t* __restrict__ resets,
    const float* __restrict__ init_h, float* __restrict__ ys) {
  __shared__ float hsh[16][H_STRIDE];                              // 16,640 B
  __shared__ __align__(16) bf16 wlds[LDS_TILES * 16 * COL_STRIDE]; // 304,128 B

  const int g    = blockIdx.x;
  const int tid  = threadIdx.x;
  const int wave = tid >> 5;
  const int lane = tid & 31;
  const int l    = lane & 15;
  const int hi   = lane >> 4;

  // ---- stage 36 weight tiles into LDS ----
  for (int u = tid; u < LDS_TILES * 16 * 16; u += 256) {
    const int tau = u >> 8;
    const int rem = u & 255;
    const int c   = rem >> 4;
    const int kb  = rem & 15;
    int col;
    if (tau < 32) {
      const int wv = tau >> 2, jj = tau & 3;
      col = (jj >> 1) * HID + 32 * wv + 16 * (jj & 1);
    } else {
      const int wv = (tau - 32) + 4;
      col = 2 * HID + 32 * wv;
    }
    const bf16* src = WhT + (size_t)(col + c) * DIM + kb * 16;
    bf16* dst = &wlds[(tau * 16 + c) * COL_STRIDE + kb * 16];
    *(int4*)dst       = *(const int4*)src;
    *(int4*)(dst + 8) = *(const int4*)(src + 8);
  }

  // ---- init hidden state ----
  for (int i = tid; i < 16 * HID; i += 256) {
    const int row = i >> 8, col = i & 255;
    hsh[row][col] = init_h[(size_t)(g * 16 + row) * HID + col];
  }

  // per-thread gate biases (folded into accumulator init)
  float br[2], bz[2], bn[2];
#pragma unroll
  for (int j = 0; j < 2; ++j) {
    const int c = 32 * wave + 16 * j + l;
    br[j] = bias[c];
    bz[j] = bias[HID + c];
    bn[j] = bias[2 * HID + c];
  }
  __syncthreads();

  const bool g4 = (wave < 4);
  const bf16* gptr4 = WhT + (size_t)(2 * HID + 32 * wave + l) * DIM + hi * 16;
  const bf16* gptr5 =
      WhT + (size_t)(2 * HID + 32 * wave + 16 + l) * DIM + hi * 16;

  for (int t = 0; t < T_STEPS; ++t) {
    const bf16* gbase = gi + ((size_t)t * 16 + g) * TILE_E;

    // issue this step's gi loads now; they complete under the GEMM
    int4 giraw[6];
#pragma unroll
    for (int j = 0; j < 2; ++j) {
      const int c = 32 * wave + 16 * j + l;
#pragma unroll
      for (int gate = 0; gate < 3; ++gate) {
        giraw[j * 3 + gate] =
            *(const int4*)(gbase + ((size_t)(gate * HID + c)) * 16 + hi * 8);
      }
    }

    // prefetch gi tile t*16+g+1, the one after this in memory (24 KB = 192 x 128B lines)
    if ((t + 1 < T_STEPS) && tid < 192) {
      const char* pf = (const char*)(gbase + TILE_E);
      __builtin_prefetch(pf + tid * 128, 0, 3);
    }

    // reset mask: one 16B load of the 16 flags; thread owns col = tid
    union { uint4 u; uint8_t b[16]; } rf;
    rf.u = *(const uint4*)(resets + (size_t)t * BATCH + g * 16);
#pragma unroll
    for (int row = 0; row < 16; ++row) {
      if (rf.b[row]) hsh[row][tid] = 0.0f;
    }
    __syncthreads();

    // recurrent GEMM; acc pre-loaded with gate biases
    v8f acc[6];
    acc[0] = splat8(br[0]); acc[1] = splat8(br[1]);
    acc[2] = splat8(bz[0]); acc[3] = splat8(bz[1]);
    acc[4] = splat8(bn[0]); acc[5] = splat8(bn[1]);

#pragma unroll 2
    for (int k0 = 0; k0 < HID; k0 += 32) {
      v16bf bg5 = load_b_frag(gptr5 + k0);
      v16bf bg4;
      if (g4) bg4 = load_b_frag(gptr4 + k0);

      v16bf af;
#pragma unroll
      for (int e = 0; e < 8; ++e) {
        af[e]     = (bf16)hsh[l][k0 + hi * 16 + e];
        af[8 + e] = (bf16)hsh[l][k0 + hi * 16 + 8 + e];
      }

#pragma unroll
      for (int j6 = 0; j6 < 4; ++j6) {
        const int tix = wave * 4 + j6;
        const bf16* bp = &wlds[(tix * 16 + l) * COL_STRIDE + k0 + hi * 16];
        acc[j6] = __builtin_amdgcn_wmma_f32_16x16x32_bf16(
            false, af, false, load_b_frag(bp), (short)0, acc[j6],
            false, false);
      }

      if (g4) {
        acc[4] = __builtin_amdgcn_wmma_f32_16x16x32_bf16(
            false, af, false, bg4, (short)0, acc[4], false, false);
      } else {
        const int tix = 32 + (wave - 4);
        const bf16* bp = &wlds[(tix * 16 + l) * COL_STRIDE + k0 + hi * 16];
        acc[4] = __builtin_amdgcn_wmma_f32_16x16x32_bf16(
            false, af, false, load_b_frag(bp), (short)0, acc[4],
            false, false);
      }

      acc[5] = __builtin_amdgcn_wmma_f32_16x16x32_bf16(
          false, af, false, bg5, (short)0, acc[5], false, false);
    }
    __syncthreads();

    // fused gates; each thread owns 16 unique (row, col) elements
#pragma unroll
    for (int j = 0; j < 2; ++j) {
      const int c = 32 * wave + 16 * j + l;
      union { int4 u; bf16 h[8]; } ur, uz, un;
      ur.u = giraw[j * 3 + 0];
      uz.u = giraw[j * 3 + 1];
      un.u = giraw[j * 3 + 2];
#pragma unroll
      for (int r = 0; r < 8; ++r) {
        const int row   = hi * 8 + r;
        const float rg  = sigmoidf((float)ur.h[r] + acc[j][r]);
        const float zg  = sigmoidf((float)uz.h[r] + acc[2 + j][r]);
        const float ng  = fast_tanh((float)un.h[r] + rg * acc[4 + j][r]);
        const float hold = hsh[row][c];
        const float nh   = (1.0f - zg) * ng + zg * hold;
        hsh[row][c] = nh;
        ys[((size_t)t * BATCH + g * 16 + row) * HID + c] = nh;
      }
    }
    __syncthreads();
  }
}

// ---------------------------------------------------------------------------
// Host launcher
// ---------------------------------------------------------------------------
extern "C" void kernel_launch(void* const* d_in, const int* in_sizes, int n_in,
                              void* d_out, int out_size, void* d_ws,
                              size_t ws_size, hipStream_t stream) {
  (void)in_sizes; (void)n_in; (void)out_size; (void)ws_size;

  const float*   ins    = (const float*)d_in[0];
  const uint8_t* rst    = (const uint8_t*)d_in[1];
  const float*   init_h = (const float*)d_in[2];
  const float*   Wi     = (const float*)d_in[3];
  const float*   Wh     = (const float*)d_in[4];
  const float*   bias   = (const float*)d_in[5];
  float* ys = (float*)d_out;

  char* ws = (char*)d_ws;
  bf16* WiT = (bf16*)(ws);
  bf16* WhT = (bf16*)(ws + 393216);
  bf16* gi  = (bf16*)(ws + 786432);

  convert_weights<<<NG, DIM, 0, stream>>>(Wi, Wh, WiT, WhT);
  input_gemm<<<12288, 256, 0, stream>>>(ins, WiT, gi);
  rnn_scan<<<16, 256, 0, stream>>>(gi, WhT, bias, rst, init_h, ys);
}
